// CoarseSkeletonHead_18236431139324
// MI455X (gfx1250) — matrix operand order repaired and checked by reading, not compiled, not run on hardware
//
#include <hip/hip_runtime.h>
#include <hip/hip_bf16.h>

typedef __attribute__((ext_vector_type(16))) __bf16 v16bf;
typedef __attribute__((ext_vector_type(8)))  float  v8f;

#define USE_ASYNC_LDS 1

union Frag {
  v16bf v;
  uint4 q[2];
  unsigned int w[8];
  unsigned short s[16];
};

__constant__ int PAR[24] = {-1,0,0,0,1,2,3,4,5,6,7,8,9,9,9,12,13,14,16,17,18,19,20,21};

__device__ __forceinline__ unsigned short f2bf(float f) {
  unsigned int u = __float_as_uint(f);
  u += 0x7FFFu + ((u >> 16) & 1u);   // round to nearest even
  return (unsigned short)(u >> 16);
}

// pack two f32 -> two bf16 in one dword (hardware v_cvt_pk_bf16_f32 when available)
#if defined(__has_builtin) && __has_builtin(__builtin_amdgcn_cvt_pk_bf16_f32)
typedef __attribute__((ext_vector_type(2))) __bf16 v2bf;
__device__ __forceinline__ unsigned int pkbf(float a, float b) {
  union { v2bf v; unsigned int u; } c;
  c.v = __builtin_amdgcn_cvt_pk_bf16_f32(a, b);
  return c.u;
}
#else
__device__ __forceinline__ unsigned int pkbf(float a, float b) {
  return (unsigned int)f2bf(a) | ((unsigned int)f2bf(b) << 16);
}
#endif

__device__ __forceinline__ void cvt8p(const float* __restrict__ p, unsigned int* w) {
  float4 a = *(const float4*)p;
  float4 b = *(const float4*)(p + 4);
  w[0] = pkbf(a.x, a.y); w[1] = pkbf(a.z, a.w);
  w[2] = pkbf(b.x, b.y); w[3] = pkbf(b.z, b.w);
}

// Cooperative 32KB global->LDS stage: 2048 x b128, 8 per thread (256 threads).
__device__ __forceinline__ void stage32k(unsigned short* dst, const unsigned short* src, int tid) {
#if USE_ASYNC_LDS
  unsigned int lbase = (unsigned int)(unsigned long long)(void*)dst;  // LDS byte offset
  const char* g = (const char*)src;
#pragma unroll
  for (int i = 0; i < 8; ++i) {
    unsigned int l = lbase + (unsigned int)(tid + i * 256) * 16u;
    const void* ga = g + (size_t)(tid + i * 256) * 16u;
    asm volatile("global_load_async_to_lds_b128 %0, %1, off"
                 :: "v"(l), "v"(ga) : "memory");
  }
  asm volatile("s_wait_asynccnt 0" ::: "memory");
#else
  const uint4* s4 = (const uint4*)src;
  uint4* d4 = (uint4*)dst;
#pragma unroll
  for (int i = 0; i < 8; ++i) d4[tid + i * 256] = s4[tid + i * 256];
#endif
}

// Convert + transpose weights to bf16:  W1t[512][256], W2t[96][512] (rows 92..95 zero)
__global__ void prep_weights(const float* __restrict__ W1, const float* __restrict__ W2,
                             unsigned short* __restrict__ w1t, unsigned short* __restrict__ w2t) {
  int i = blockIdx.x * blockDim.x + threadIdx.x;
  if (i < 512*256) {
    int h = i >> 8, d = i & 255;
    w1t[i] = f2bf(W1[d*512 + h]);
  }
  int j = i - 512*256;
  if (j >= 0 && j < 96*512) {
    int n = j >> 9, k = j & 511;
    w2t[j] = (n < 92) ? f2bf(W2[k*92 + n]) : (unsigned short)0;
  }
}

#define BM 128

__global__ __launch_bounds__(256) void mlp_skeleton(
    const float* __restrict__ z, const float* __restrict__ b1v, const float* __restrict__ b2v,
    const unsigned short* __restrict__ w1t, const unsigned short* __restrict__ w2t,
    float* __restrict__ out, int Btot)
{
  // 128KB h-tile region (later reused: raw[128][96] f32 @0, off[128][72] f32 @49152) + 32KB stage
  __shared__ __align__(16) unsigned char smem[163840];
  unsigned short* hT    = (unsigned short*)smem;               // [128][512] bf16
  float*          rawT  = (float*)smem;                        // [128][96]
  float*          offT  = (float*)(smem + 49152);              // [128][72]
  unsigned short* stage = (unsigned short*)(smem + 131072);    // 32KB staging

  const int tid   = threadIdx.x;
  const int wave  = tid >> 5;
  const int lane  = tid & 31;
  const int lmod  = lane & 15;
  const int lhalf = lane >> 4;
  const long grow0 = (long)blockIdx.x * BM;

  const v8f zero8 = {0.f,0.f,0.f,0.f,0.f,0.f,0.f,0.f};

  // ---- A fragments of z (this wave's 16 rows, full K=256) -> bf16, kept in regs ----
  Frag afrag[8];
  {
    const float* zrow = z + (grow0 + wave*16 + lmod) * 256;
#pragma unroll
    for (int kt = 0; kt < 8; ++kt) {
      int c = kt*32 + lhalf*8;
      cvt8p(zrow + c,      afrag[kt].w);
      cvt8p(zrow + c + 16, afrag[kt].w + 4);
    }
  }

  // ---- GEMM1: h = silu(z @ W1 + b1) -> LDS bf16. 8 rounds x 4 n-tiles (64 cols) ----
  for (int rt = 0; rt < 8; ++rt) {
    __syncthreads();
    stage32k(stage, w1t + rt*64*256, tid);   // 4 tiles x [16][256] bf16 = 32KB
    __syncthreads();
#pragma unroll
    for (int sub = 0; sub < 4; ++sub) {
      int nt = rt*4 + sub;
      v8f acc = zero8;
#pragma unroll
      for (int kt = 0; kt < 8; ++kt) {
        Frag bf_;
        const uint4* p = (const uint4*)(stage + (sub*16 + lmod)*256 + kt*32 + lhalf*8);
        bf_.q[0] = p[0]; bf_.q[1] = p[2];
        acc = __builtin_amdgcn_wmma_f32_16x16x32_bf16(false, afrag[kt].v, false, bf_.v,
                                                      (short)0, acc, false, false);
      }
      int col = nt*16 + lmod;
      float bias = b1v[col];
#pragma unroll
      for (int v = 0; v < 8; v += 2) {
        int lr = wave*16 + v + 8*lhalf;
        float x0 = acc[v]   + bias;
        float x1 = acc[v+1] + bias;
        float s0 = x0 * __builtin_amdgcn_rcpf(1.0f + __expf(-x0));  // silu
        float s1 = x1 * __builtin_amdgcn_rcpf(1.0f + __expf(-x1));
        unsigned int pk = pkbf(s0, s1);
        hT[lr*512 + col]     = (unsigned short)(pk & 0xffffu);
        hT[(lr+1)*512 + col] = (unsigned short)(pk >> 16);
      }
    }
  }

  // ---- GEMM2: raw = h @ W2 + b2 (92 -> 96 cols, 6 n-tiles). 3 rounds x 2 n-tiles ----
  v8f cacc[6];
#pragma unroll
  for (int t = 0; t < 6; ++t) cacc[t] = zero8;

#pragma unroll
  for (int rt = 0; rt < 3; ++rt) {
    __syncthreads();
    stage32k(stage, w2t + rt*32*512, tid);   // 2 tiles x [16][512] bf16 = 32KB
    __syncthreads();
#pragma unroll
    for (int sub = 0; sub < 2; ++sub) {
      int nt2 = rt*2 + sub;
#pragma unroll
      for (int kt = 0; kt < 16; ++kt) {
        Frag a2, bf_;
        const unsigned short* hp = hT + (wave*16 + lmod)*512 + kt*32 + lhalf*8;
        a2.q[0] = *(const uint4*)hp;
        a2.q[1] = *(const uint4*)(hp + 16);
        const uint4* p = (const uint4*)(stage + (sub*16 + lmod)*512 + kt*32 + lhalf*8);
        bf_.q[0] = p[0]; bf_.q[1] = p[2];
        cacc[nt2] = __builtin_amdgcn_wmma_f32_16x16x32_bf16(false, a2.v, false, bf_.v,
                                                            (short)0, cacc[nt2], false, false);
      }
    }
  }
  __syncthreads();   // all waves done reading hT; safe to overlay with rawT

  // spill raw tile (f32) to LDS so any thread can post-process any (row, joint)
#pragma unroll
  for (int nt2 = 0; nt2 < 6; ++nt2) {
    int col = nt2*16 + lmod;
    if (col < 92) {
      float bias = b2v[col];
#pragma unroll
      for (int v = 0; v < 8; ++v) {
        int lr = wave*16 + v + 8*lhalf;
        rawT[lr*96 + col] = cacc[nt2][v] + bias;
      }
    }
  }
  __syncthreads();

  float* joints_out  = out;
  float* offsets_out = out + (long)Btot*72;
  float* length_out  = out + (long)Btot*144;

  // root joint offsets = 0
  for (int i = tid; i < BM; i += 256) {
    long gr = grow0 + i;
    offsets_out[gr*72+0]=0.f; offsets_out[gr*72+1]=0.f; offsets_out[gr*72+2]=0.f;
    offT[i*72+0]=0.f; offT[i*72+1]=0.f; offT[i*72+2]=0.f;
  }
  // non-root: normalize direction, softplus length, offset = dir*len
  for (int i = tid; i < BM*23; i += 256) {
    int lr = i / 23, jm1 = i % 23, j = jm1 + 1;
    const float* rp = rawT + lr*96 + jm1*4;
    float d0 = rp[0], d1 = rp[1], d2 = rp[2], lraw = rp[3];
    float nrm = sqrtf(d0*d0 + d1*d1 + d2*d2);
    float inv = __builtin_amdgcn_rcpf(fmaxf(nrm, 1e-6f));
    float len = (lraw > 20.0f) ? lraw : log1pf(__expf(lraw));
    float o0 = d0*inv*len, o1 = d1*inv*len, o2 = d2*inv*len;
    long gr = grow0 + lr;
    offsets_out[gr*72 + j*3 + 0] = o0;
    offsets_out[gr*72 + j*3 + 1] = o1;
    offsets_out[gr*72 + j*3 + 2] = o2;
    length_out[gr*23 + jm1] = len;
    offT[lr*72 + j*3 + 0] = o0;
    offT[lr*72 + j*3 + 1] = o1;
    offT[lr*72 + j*3 + 2] = o2;
  }
  __syncthreads();
  // joints = ancestor-chain prefix sums of offsets
  for (int i = tid; i < BM*24; i += 256) {
    int lr = i / 24, j = i % 24;
    float s0 = 0.f, s1 = 0.f, s2 = 0.f;
    int k = j;
    while (k > 0) {
      s0 += offT[lr*72 + k*3 + 0];
      s1 += offT[lr*72 + k*3 + 1];
      s2 += offT[lr*72 + k*3 + 2];
      k = PAR[k];
    }
    long gr = grow0 + lr;
    joints_out[gr*72 + j*3 + 0] = s0;
    joints_out[gr*72 + j*3 + 1] = s1;
    joints_out[gr*72 + j*3 + 2] = s2;
  }
}

extern "C" void kernel_launch(void* const* d_in, const int* in_sizes, int n_in,
                              void* d_out, int out_size, void* d_ws, size_t ws_size,
                              hipStream_t stream) {
  const float* z  = (const float*)d_in[0];
  const float* W1 = (const float*)d_in[1];
  const float* b1 = (const float*)d_in[2];
  const float* W2 = (const float*)d_in[3];
  const float* b2 = (const float*)d_in[4];
  int Btot = in_sizes[0] / 256;

  unsigned short* w1t = (unsigned short*)d_ws;                       // 512*256 bf16 = 256KB
  unsigned short* w2t = (unsigned short*)((char*)d_ws + 512*256*2);  //  96*512 bf16 =  96KB

  int prepN = 512*256 + 96*512;
  prep_weights<<<(prepN + 255) / 256, 256, 0, stream>>>(W1, W2, w1t, w2t);
  mlp_skeleton<<<Btot / BM, 256, 0, stream>>>(z, b1, b2, w1t, w2t, (float*)d_out, Btot);
}
